// AttentiveFPModel_37443524887200
// MI455X (gfx1250) — matrix operand order repaired and checked by reading, not compiled, not run on hardware
//
#include <hip/hip_runtime.h>
#include <hip/hip_bf16.h>
#include <cmath>

// ---------------- problem constants (match reference) ----------------
constexpr int NN = 50000;    // nodes
constexpr int EE = 500000;   // edges
constexpr int BB = 2048;     // graphs
constexpr unsigned NEG_INF_ENC = 0x007FFFFFu; // fenc(-inf)

typedef __attribute__((ext_vector_type(16))) _Float16 v16h;
typedef __attribute__((ext_vector_type(8)))  _Float16 v8h;
typedef __attribute__((ext_vector_type(8)))  float    v8f;

// ------- ordered-uint encoding of float for atomicMax-based segment max -------
__device__ __forceinline__ unsigned fenc(float f) {
  unsigned u = __float_as_uint(f);
  return (u & 0x80000000u) ? ~u : (u | 0x80000000u);
}
__device__ __forceinline__ float fdec(unsigned u) {
  return __uint_as_float((u & 0x80000000u) ? (u ^ 0x80000000u) : ~u);
}
// branchless leaky-relu, slope 0.01 (v>=0: v >= 0.01v ; v<0: 0.01v > v)
__device__ __forceinline__ float leaky(float v) { return fmaxf(v, 0.01f * v); }
__device__ __forceinline__ float eluf(float v)  { return v > 0.f ? v : (expf(v) - 1.f); }

// =====================================================================
// Generic WMMA GEMM:  Y[M,NOUT] = epi( X[M,64] @ W[NOUT,64(ldw)]^T + bias )
// EPI: 0=none 1=leaky(0.01) 2=relu 3=elu
// 128 threads = 4 waves; each wave computes 16 rows x NOUT cols via
// v_wmma_f32_16x16x32_f16 (two K-chunks of 32). Full-tile fast path avoids
// per-element predication except in the single ragged tail block.
// =====================================================================
template <int NOUT, int EPI, bool HB>
__global__ __launch_bounds__(128) void k_gemm(
    const float* __restrict__ X, const float* __restrict__ W, int ldw,
    const float* __restrict__ bias, float* __restrict__ Y, int M)
{
  __shared__ __align__(16) _Float16 Ws[NOUT * 72];
  __shared__ __align__(16) _Float16 Xs[64 * 72];
  const int tid  = threadIdx.x;
  const int row0 = blockIdx.x * 64;
  const bool full = (row0 + 64 <= M);

  for (int i = tid; i < NOUT * 64; i += 128) {
    int n = i >> 6, k = i & 63;
    Ws[n * 72 + k] = (_Float16)W[n * ldw + k];
  }
  for (int i = tid; i < 64 * 64; i += 128) {
    int r = i >> 6, k = i & 63;
    int gr = row0 + r;
    Xs[r * 72 + k] = (full || gr < M) ? (_Float16)X[(size_t)gr * 64 + k] : (_Float16)0.f;
  }
  __syncthreads();

  const int wave = tid >> 5, lane = tid & 31;
  const int lr = lane & 15, hi = lane >> 4;

  // A fragment (16x32 f16 layout: lanes 0-15 K{0..7,16..23}, lanes 16-31 K{8..15,24..31})
  const _Float16* xrow = &Xs[(wave * 16 + lr) * 72];
  v8h a0l = *(const v8h*)(xrow + hi * 8);
  v8h a0h = *(const v8h*)(xrow + 16 + hi * 8);
  v8h a1l = *(const v8h*)(xrow + 32 + hi * 8);
  v8h a1h = *(const v8h*)(xrow + 48 + hi * 8);
  v16h A0, A1;
#pragma unroll
  for (int i = 0; i < 8; ++i) {
    A0[i] = a0l[i]; A0[i + 8] = a0h[i];
    A1[i] = a1l[i]; A1[i + 8] = a1h[i];
  }

#pragma unroll
  for (int nt = 0; nt < NOUT / 16; ++nt) {
    // B fragment (32x16): lanes 0-15 K{0..7,16..23}, lanes 16-31 K{8..15,24..31} (matches A)
    const _Float16* wrow = &Ws[(nt * 16 + lr) * 72];
    v8h b0l = *(const v8h*)(wrow + hi * 8);
    v8h b0h = *(const v8h*)(wrow + 16 + hi * 8);
    v8h b1l = *(const v8h*)(wrow + 32 + hi * 8);
    v8h b1h = *(const v8h*)(wrow + 48 + hi * 8);
    v16h B0, B1;
#pragma unroll
    for (int i = 0; i < 8; ++i) {
      B0[i] = b0l[i]; B0[i + 8] = b0h[i];
      B1[i] = b1l[i]; B1[i + 8] = b1h[i];
    }
    v8f acc = {};
    acc = __builtin_amdgcn_wmma_f32_16x16x32_f16(false, A0, false, B0, (short)0, acc, false, false);
    acc = __builtin_amdgcn_wmma_f32_16x16x32_f16(false, A1, false, B1, (short)0, acc, false, false);

    const int col = nt * 16 + lr;
    const float bv = HB ? bias[col] : 0.f;
    const int rbase = row0 + wave * 16 + hi * 8;  // D rows rbase..rbase+7
    float* __restrict__ yp = Y + (size_t)rbase * NOUT + col;
    if (full) {
#pragma unroll
      for (int i = 0; i < 8; ++i) {
        float v = acc[i] + bv;
        if      (EPI == 1) v = fmaxf(v, 0.01f * v);
        else if (EPI == 2) v = fmaxf(v, 0.f);
        else if (EPI == 3) v = eluf(v);
        yp[(size_t)i * NOUT] = v;
      }
    } else {
#pragma unroll
      for (int i = 0; i < 8; ++i) {
        if (rbase + i < M) {
          float v = acc[i] + bv;
          if      (EPI == 1) v = fmaxf(v, 0.01f * v);
          else if (EPI == 2) v = fmaxf(v, 0.f);
          else if (EPI == 3) v = eluf(v);
          yp[(size_t)i * NOUT] = v;
        }
      }
    }
  }
}

// ---------------- fills ----------------
__global__ void k_fill_f32(float* p, float v, int n) {
  int i = blockIdx.x * 256 + threadIdx.x;
  if (i < n) p[i] = v;
}
__global__ void k_fill_u32(unsigned* p, unsigned v, int n) {
  int i = blockIdx.x * 256 + threadIdx.x;
  if (i < n) p[i] = v;
}
__global__ void k_relu_ip(float* p, int n) {
  int i = blockIdx.x * 256 + threadIdx.x;
  if (i < n) p[i] = fmaxf(p[i], 0.f);
}

// ---------------- per-row dot products (warp per row) ----------------
__global__ __launch_bounds__(256) void k_rowdot(const float* __restrict__ X,
    const float* __restrict__ v, float* __restrict__ d, int M) {
  int r = blockIdx.x * 8 + (threadIdx.x >> 5);
  if (r >= M) return;
  int lane = threadIdx.x & 31;
  float a = X[(size_t)r * 64 + lane] * v[lane] + X[(size_t)r * 64 + lane + 32] * v[lane + 32];
#pragma unroll
  for (int off = 16; off > 0; off >>= 1) a += __shfl_xor(a, off, 32);
  if (lane == 0) d[r] = a;
}
__global__ __launch_bounds__(256) void k_rowdot2(const float* __restrict__ X,
    const float* __restrict__ v1, const float* __restrict__ v2,
    float* __restrict__ d1, float* __restrict__ d2, int M) {
  int r = blockIdx.x * 8 + (threadIdx.x >> 5);
  if (r >= M) return;
  int lane = threadIdx.x & 31;
  float xa = X[(size_t)r * 64 + lane], xb = X[(size_t)r * 64 + lane + 32];
  float a = xa * v1[lane] + xb * v1[lane + 32];
  float b = xa * v2[lane] + xb * v2[lane + 32];
#pragma unroll
  for (int off = 16; off > 0; off >>= 1) { a += __shfl_xor(a, off, 32); b += __shfl_xor(b, off, 32); }
  if (lane == 0) { d1[r] = a; d2[r] = b; }
}

// ---------------- GATEConv edge logits (warp per edge) ----------------
// t = leaky(p[src] + ea @ W1b^T); logit = leaky(dot(t, att_l) + qr[dst])
__global__ __launch_bounds__(256) void k_gate_logits(
    const float* __restrict__ p, const float* __restrict__ qr,
    const float* __restrict__ ea, const float* __restrict__ W1,   // gate_lin1_w [64][80]
    const float* __restrict__ attl,
    const int* __restrict__ src, const int* __restrict__ dst,
    float* __restrict__ elog, unsigned* __restrict__ nmax)
{
  __shared__ float sW[64 * 16];
  __shared__ float sA[64];
  int tid = threadIdx.x;
  for (int i = tid; i < 64 * 16; i += 256) { int f = i >> 4, k = i & 15; sW[i] = W1[f * 80 + 64 + k]; }
  if (tid < 64) sA[tid] = attl[tid];
  __syncthreads();

  int e = blockIdx.x * 8 + (tid >> 5);
  if (e >= EE) return;
  int lane = tid & 31;
  int s = src[e], d = dst[e];
  float acc = 0.f;
#pragma unroll
  for (int half = 0; half < 2; ++half) {
    int f = lane + half * 32;
    float t = p[(size_t)s * 64 + f];
#pragma unroll
    for (int k = 0; k < 16; ++k) t += ea[(size_t)e * 16 + k] * sW[f * 16 + k];
    t = leaky(t);
    acc += t * sA[f];
  }
#pragma unroll
  for (int off = 16; off > 0; off >>= 1) acc += __shfl_xor(acc, off, 32);
  if (lane == 0) {
    float lg = leaky(acc + qr[d]);
    elog[e] = lg;
    atomicMax(&nmax[d], fenc(lg));
  }
}

// ---------------- GATConv edge logits (thread per edge) ----------------
__global__ void k_gat_logits(const float* __restrict__ asrc, const float* __restrict__ adst,
    const int* __restrict__ src, const int* __restrict__ dst,
    float* __restrict__ elog, unsigned* __restrict__ nmax) {
  int e = blockIdx.x * 256 + threadIdx.x;
  if (e >= EE) return;
  float lg = leaky(asrc[src[e]] + adst[dst[e]]);
  elog[e] = lg;
  atomicMax(&nmax[dst[e]], fenc(lg));
}

// ---------------- segment softmax passes over edges ----------------
__global__ void k_edge_exp(float* __restrict__ elog, const unsigned* __restrict__ nmax,
    const int* __restrict__ dst, float* __restrict__ nsum) {
  int e = blockIdx.x * 256 + threadIdx.x;
  if (e >= EE) return;
  int d = dst[e];
  float w = expf(elog[e] - fdec(nmax[d]));
  elog[e] = w;
  atomicAdd(&nsum[d], w);
}
__global__ __launch_bounds__(256) void k_edge_agg(const float* __restrict__ elog,
    const float* __restrict__ nsum, const float* __restrict__ feat,
    const int* __restrict__ src, const int* __restrict__ dst, float* __restrict__ agg) {
  int e = blockIdx.x * 8 + (threadIdx.x >> 5);
  if (e >= EE) return;
  int lane = threadIdx.x & 31;
  int s = src[e], d = dst[e];
  float alpha = elog[e] / (nsum[d] + 1e-16f);
  atomicAdd(&agg[(size_t)d * 64 + lane],      alpha * feat[(size_t)s * 64 + lane]);
  atomicAdd(&agg[(size_t)d * 64 + lane + 32], alpha * feat[(size_t)s * 64 + lane + 32]);
}

// ---------------- elementwise: bias + ELU ----------------
__global__ void k_bias_elu(const float* __restrict__ in, const float* __restrict__ bias,
    float* __restrict__ out, int M) {
  int i = blockIdx.x * 256 + threadIdx.x;
  if (i >= M * 64) return;
  out[i] = eluf(in[i] + bias[i & 63]);
}

// ---------------- GRU combine (biases already in gx/gh) ----------------
__global__ void k_gru(const float* __restrict__ gx, const float* __restrict__ gh,
    const float* __restrict__ hprev, float* __restrict__ out, int M) {
  int i = blockIdx.x * 256 + threadIdx.x;
  if (i >= M * 64) return;
  int r = i >> 6, f = i & 63;
  size_t b = (size_t)r * 192;
  float xr = gx[b + f], xz = gx[b + 64 + f], xn = gx[b + 128 + f];
  float hr = gh[b + f], hz = gh[b + 64 + f], hn = gh[b + 128 + f];
  float rr = 1.f / (1.f + expf(-(xr + hr)));
  float zz = 1.f / (1.f + expf(-(xz + hz)));
  float nn = tanhf(xn + rr * hn);
  float h  = hprev[i];
  out[i] = fmaxf((1.f - zz) * nn + zz * h, 0.f);
}

// ---------------- node -> batch scatter sum (warp per node) ----------------
__global__ __launch_bounds__(256) void k_node2batch(const float* __restrict__ x,
    const int* __restrict__ batch, float* __restrict__ outB) {
  int n = blockIdx.x * 8 + (threadIdx.x >> 5);
  if (n >= NN) return;
  int lane = threadIdx.x & 31, b = batch[n];
  atomicAdd(&outB[(size_t)b * 64 + lane],      x[(size_t)n * 64 + lane]);
  atomicAdd(&outB[(size_t)b * 64 + lane + 32], x[(size_t)n * 64 + lane + 32]);
}

// ---------------- mol readout softmax over batch segments ----------------
__global__ void k_mol_logits(const float* __restrict__ asrc, const float* __restrict__ gB,
    const int* __restrict__ batch, float* __restrict__ nlog, unsigned* __restrict__ bmax) {
  int n = blockIdx.x * 256 + threadIdx.x;
  if (n >= NN) return;
  float lg = leaky(asrc[n] + gB[batch[n]]);
  nlog[n] = lg;
  atomicMax(&bmax[batch[n]], fenc(lg));
}
__global__ void k_mol_exp(float* __restrict__ nlog, const unsigned* __restrict__ bmax,
    const int* __restrict__ batch, float* __restrict__ bsum) {
  int n = blockIdx.x * 256 + threadIdx.x;
  if (n >= NN) return;
  int b = batch[n];
  float w = expf(nlog[n] - fdec(bmax[b]));
  nlog[n] = w;
  atomicAdd(&bsum[b], w);
}
__global__ __launch_bounds__(256) void k_mol_agg(const float* __restrict__ nlog,
    const float* __restrict__ bsum, const float* __restrict__ xs,
    const int* __restrict__ batch, float* __restrict__ aggB) {
  int n = blockIdx.x * 8 + (threadIdx.x >> 5);
  if (n >= NN) return;
  int lane = threadIdx.x & 31, b = batch[n];
  float alpha = nlog[n] / (bsum[b] + 1e-16f);
  atomicAdd(&aggB[(size_t)b * 64 + lane],      alpha * xs[(size_t)n * 64 + lane]);
  atomicAdd(&aggB[(size_t)b * 64 + lane + 32], alpha * xs[(size_t)n * 64 + lane + 32]);
}

// ---------------- regression head: LN -> 64 -> 64 -> 1 (block per graph) ----------------
__global__ __launch_bounds__(64) void k_head(const float* __restrict__ emb,
    const float* __restrict__ ln_g, const float* __restrict__ ln_b,
    const float* __restrict__ w1, const float* __restrict__ b1,
    const float* __restrict__ w2, const float* __restrict__ b2,
    const float* __restrict__ w3, const float* __restrict__ b3,
    float* __restrict__ out) {
  int b = blockIdx.x, t = threadIdx.x;
  __shared__ float s[64], y[64], z[64];
  float e = emb[(size_t)b * 64 + t];
  s[t] = e; __syncthreads();
  for (int o = 32; o > 0; o >>= 1) { if (t < o) s[t] += s[t + o]; __syncthreads(); }
  float mu = s[0] / 64.f; __syncthreads();
  float d = e - mu;
  s[t] = d * d; __syncthreads();
  for (int o = 32; o > 0; o >>= 1) { if (t < o) s[t] += s[t + o]; __syncthreads(); }
  float var = s[0] / 64.f; __syncthreads();
  y[t] = d * rsqrtf(var + 1e-5f) * ln_g[t] + ln_b[t]; __syncthreads();
  float a = b1[t];
  for (int k = 0; k < 64; ++k) a += w1[t * 64 + k] * y[k];
  z[t] = fmaxf(a, 0.f); __syncthreads();
  a = b2[t];
  for (int k = 0; k < 64; ++k) a += w2[t * 64 + k] * z[k];
  float h2v = fmaxf(a, 0.f);
  s[t] = h2v * w3[t]; __syncthreads();
  for (int o = 32; o > 0; o >>= 1) { if (t < o) s[t] += s[t + o]; __syncthreads(); }
  if (t == 0) out[b] = s[0] + b3[0];
}

// =====================================================================
extern "C" void kernel_launch(void* const* d_in, const int* in_sizes, int n_in,
                              void* d_out, int out_size, void* d_ws, size_t ws_size,
                              hipStream_t stream) {
  (void)in_sizes; (void)n_in; (void)out_size; (void)ws_size;
  // ---- inputs (setup_inputs order) ----
  const float* x_in        = (const float*)d_in[0];
  const float* edge_attr   = (const float*)d_in[1];
  const float* lin1_w      = (const float*)d_in[2];
  const float* lin1_b      = (const float*)d_in[3];
  const float* gate_lin1_w = (const float*)d_in[4];
  const float* gate_lin2_w = (const float*)d_in[5];
  const float* gate_att_l  = (const float*)d_in[6];
  const float* gate_att_r  = (const float*)d_in[7];
  const float* gate_bias   = (const float*)d_in[8];
  const float* gru0_wx     = (const float*)d_in[9];
  const float* gru0_wh     = (const float*)d_in[10];
  const float* gru0_bx     = (const float*)d_in[11];
  const float* gru0_bh     = (const float*)d_in[12];
  const float* atom_lin_w  = (const float*)d_in[13];
  const float* atom_att_s  = (const float*)d_in[14];
  const float* atom_att_d  = (const float*)d_in[15];
  const float* atom_bias   = (const float*)d_in[16];
  const float* atom_gru_wx = (const float*)d_in[17];
  const float* atom_gru_wh = (const float*)d_in[18];
  const float* atom_gru_bx = (const float*)d_in[19];
  const float* atom_gru_bh = (const float*)d_in[20];
  const float* mol_lin_w   = (const float*)d_in[21];
  const float* mol_att_src = (const float*)d_in[22];
  const float* mol_att_dst = (const float*)d_in[23];
  const float* mol_bias    = (const float*)d_in[24];
  const float* mol_gru_wx  = (const float*)d_in[25];
  const float* mol_gru_wh  = (const float*)d_in[26];
  const float* mol_gru_bx  = (const float*)d_in[27];
  const float* mol_gru_bh  = (const float*)d_in[28];
  const float* lin2_w      = (const float*)d_in[29];
  const float* lin2_b      = (const float*)d_in[30];
  const float* ln_g        = (const float*)d_in[31];
  const float* ln_b        = (const float*)d_in[32];
  const float* h1_w        = (const float*)d_in[33];
  const float* h1_b        = (const float*)d_in[34];
  const float* h2_w        = (const float*)d_in[35];
  const float* h2_b        = (const float*)d_in[36];
  const float* h3_w        = (const float*)d_in[37];
  const float* h3_b        = (const float*)d_in[38];
  const int*   edge_index  = (const int*)d_in[39];
  const int*   batch       = (const int*)d_in[40];
  const int* src = edge_index;
  const int* dst = edge_index + EE;
  float* outP = (float*)d_out;

  // ---- workspace bump allocator ----
  float* w = (float*)d_ws;
  size_t off = 0;
  auto alloc = [&](size_t n) { float* p = w + off; off += n; return p; };
  float* xh    = alloc((size_t)NN * 64);   // node state
  float* pbuf  = alloc((size_t)NN * 64);   // gate p / xl / xs
  float* xin   = alloc((size_t)NN * 64);   // elu(h)
  float* agg   = alloc((size_t)NN * 64);   // softmax-weighted aggregation
  float* gx    = alloc((size_t)NN * 192);
  float* gh    = alloc((size_t)NN * 192);
  float* elog  = alloc((size_t)EE);
  float* nmaxf = alloc((size_t)NN);        // encoded uint max
  float* nsum  = alloc((size_t)NN);
  float* dN1   = alloc((size_t)NN);        // qr / asrc / a_src(mol)
  float* dN2   = alloc((size_t)NN);        // adst / mol per-node logits
  float* outB  = alloc((size_t)BB * 64);   // readout state
  float* aggB  = alloc((size_t)BB * 64);
  float* proj  = alloc((size_t)BB * 64);   // out @ mol_lin / emb
  float* bmaxf = alloc((size_t)BB);
  float* bsum  = alloc((size_t)BB);
  float* gB    = alloc((size_t)BB);
  unsigned* nmax = (unsigned*)nmaxf;
  unsigned* bmax = (unsigned*)bmaxf;

  const int gN   = (NN + 63) / 64;            // GEMM grid for node-rows
  const int gBm  = (BB + 63) / 64;            // GEMM grid for graph-rows
  const int eW   = (EE + 7) / 8;              // warp-per-edge blocks
  const int nW   = (NN + 7) / 8;              // warp-per-node blocks
  const int eT   = (EE + 255) / 256;          // thread-per-edge
  const int nT   = (NN + 255) / 256;          // thread-per-node
  auto eb = [](int n) { return (n + 255) / 256; };

  // ===== 1) x = leaky_relu(x @ lin1_w^T + b) =====
  k_gemm<64, 1, true><<<gN, 128, 0, stream>>>(x_in, lin1_w, 64, lin1_b, xh, NN);

  // ===== 2) GATEConv =====
  k_gemm<64, 0, false><<<gN, 128, 0, stream>>>(xh, gate_lin1_w, 80, nullptr, pbuf, NN); // p = xh @ W1a^T
  k_rowdot<<<nW, 256, 0, stream>>>(xh, gate_att_r, dN1, NN);                            // qr
  k_fill_u32<<<eb(NN), 256, 0, stream>>>(nmax, NEG_INF_ENC, NN);
  k_fill_f32<<<eb(NN), 256, 0, stream>>>(nsum, 0.f, NN);
  k_fill_f32<<<eb(NN * 64), 256, 0, stream>>>(agg, 0.f, NN * 64);
  k_gate_logits<<<eW, 256, 0, stream>>>(pbuf, dN1, edge_attr, gate_lin1_w, gate_att_l, src, dst, elog, nmax);
  k_edge_exp<<<eT, 256, 0, stream>>>(elog, nmax, dst, nsum);
  k_edge_agg<<<eW, 256, 0, stream>>>(elog, nsum, xh, src, dst, agg);
  // xin = elu(agg @ W2^T + gate_bias)
  k_gemm<64, 3, true><<<gN, 128, 0, stream>>>(agg, gate_lin2_w, 64, gate_bias, xin, NN);
  // GRU0
  k_gemm<192, 0, true><<<gN, 128, 0, stream>>>(xin, gru0_wx, 64, gru0_bx, gx, NN);
  k_gemm<192, 0, true><<<gN, 128, 0, stream>>>(xh,  gru0_wh, 64, gru0_bh, gh, NN);
  k_gru<<<eb(NN * 64), 256, 0, stream>>>(gx, gh, xh, xh, NN);

  // ===== 3) two GATConv + GRU layers =====
  for (int l = 0; l < 2; ++l) {
    const float* alw = atom_lin_w + (size_t)l * 64 * 64;
    k_gemm<64, 0, false><<<gN, 128, 0, stream>>>(xh, alw, 64, nullptr, pbuf, NN); // xl
    k_rowdot2<<<nW, 256, 0, stream>>>(pbuf, atom_att_s + l * 64, atom_att_d + l * 64, dN1, dN2, NN);
    k_fill_u32<<<eb(NN), 256, 0, stream>>>(nmax, NEG_INF_ENC, NN);
    k_fill_f32<<<eb(NN), 256, 0, stream>>>(nsum, 0.f, NN);
    k_fill_f32<<<eb(NN * 64), 256, 0, stream>>>(agg, 0.f, NN * 64);
    k_gat_logits<<<eT, 256, 0, stream>>>(dN1, dN2, src, dst, elog, nmax);
    k_edge_exp<<<eT, 256, 0, stream>>>(elog, nmax, dst, nsum);
    k_edge_agg<<<eW, 256, 0, stream>>>(elog, nsum, pbuf, src, dst, agg);
    k_bias_elu<<<eb(NN * 64), 256, 0, stream>>>(agg, atom_bias + l * 64, xin, NN);
    k_gemm<192, 0, true><<<gN, 128, 0, stream>>>(xin, atom_gru_wx + (size_t)l * 192 * 64, 64,
                                                 atom_gru_bx + l * 192, gx, NN);
    k_gemm<192, 0, true><<<gN, 128, 0, stream>>>(xh,  atom_gru_wh + (size_t)l * 192 * 64, 64,
                                                 atom_gru_bh + l * 192, gh, NN);
    k_gru<<<eb(NN * 64), 256, 0, stream>>>(gx, gh, xh, xh, NN);
  }

  // ===== 4) attentive readout =====
  k_fill_f32<<<eb(BB * 64), 256, 0, stream>>>(outB, 0.f, BB * 64);
  k_node2batch<<<nW, 256, 0, stream>>>(xh, batch, outB);
  k_relu_ip<<<eb(BB * 64), 256, 0, stream>>>(outB, BB * 64);
  k_gemm<64, 0, false><<<gN, 128, 0, stream>>>(xh, mol_lin_w, 64, nullptr, pbuf, NN); // xs
  k_rowdot<<<nW, 256, 0, stream>>>(pbuf, mol_att_src, dN1, NN);                       // a_src

  for (int t = 0; t < 3; ++t) {
    k_gemm<64, 0, false><<<gBm, 128, 0, stream>>>(outB, mol_lin_w, 64, nullptr, proj, BB);
    k_rowdot<<<(BB + 7) / 8, 256, 0, stream>>>(proj, mol_att_dst, gB, BB);
    k_fill_u32<<<eb(BB), 256, 0, stream>>>(bmax, NEG_INF_ENC, BB);
    k_fill_f32<<<eb(BB), 256, 0, stream>>>(bsum, 0.f, BB);
    k_fill_f32<<<eb(BB * 64), 256, 0, stream>>>(aggB, 0.f, BB * 64);
    k_mol_logits<<<nT, 256, 0, stream>>>(dN1, gB, batch, dN2, bmax);
    k_mol_exp<<<nT, 256, 0, stream>>>(dN2, bmax, batch, bsum);
    k_mol_agg<<<nW, 256, 0, stream>>>(dN2, bsum, pbuf, batch, aggB);
    k_bias_elu<<<eb(BB * 64), 256, 0, stream>>>(aggB, mol_bias, aggB, BB);
    k_gemm<192, 0, true><<<gBm, 128, 0, stream>>>(aggB, mol_gru_wx, 64, mol_gru_bx, gx, BB);
    k_gemm<192, 0, true><<<gBm, 128, 0, stream>>>(outB, mol_gru_wh, 64, mol_gru_bh, gh, BB);
    k_gru<<<eb(BB * 64), 256, 0, stream>>>(gx, gh, outB, outB, BB);
  }

  // ===== 5) emb + regression head =====
  k_gemm<64, 0, true><<<gBm, 128, 0, stream>>>(outB, lin2_w, 64, lin2_b, proj, BB);
  k_head<<<BB, 64, 0, stream>>>(proj, ln_g, ln_b, h1_w, h1_b, h2_w, h2_b, h3_w, h3_b, outP);
}
